// dualEncoder_6038724018676
// MI455X (gfx1250) — matrix operand order repaired and checked by reading, not compiled, not run on hardware
//
#include <hip/hip_runtime.h>
#include <math.h>

#define Bsz   8
#define Tt    12
#define Nn    512
#define Ff    128
#define Kadj  16
#define NSAMP 36
#define Mtok  (Bsz*Tt*Nn)          // 49152 tokens
#define INV_SQD 0.08838834764831845f

typedef __attribute__((ext_vector_type(16))) __bf16 v16bf;
typedef __attribute__((ext_vector_type(8)))  __bf16 v8bf;
typedef __attribute__((ext_vector_type(8)))  float  v8f;

// ---------------------------------------------------------------------------
// Weight pre-conversion: dst[n*128+k] = (bf16) src[off + k*ldbk + n*ldbn]
// ---------------------------------------------------------------------------
__global__ __launch_bounds__(256) void k_wcvt(const float* __restrict__ src,
                                              __bf16* __restrict__ dst,
                                              int ldbk, int ldbn, int off)
{
  int i = blockIdx.x * 256 + threadIdx.x;   // i = n*128 + k  (16384 total)
  int n = i >> 7, k = i & 127;
  dst[i] = (__bf16)src[off + k * ldbk + n * ldbn];
}

// ---------------------------------------------------------------------------
// bf16 WMMA GEMM:  Out[m][n] = sum_k A[m][k] * WT[n][k]  (+bias)(+Cin)(relu)
// Epilogue variants are compile-time to avoid per-element branches.
// Block: 8 waves, 16 rows x 128 cols. A tile staged in LDS as bf16.
// ---------------------------------------------------------------------------
template <int BIAS, int CIN, int RELU>
__global__ __launch_bounds__(256) void k_gemm_wmma(
    const float* __restrict__ A, const __bf16* __restrict__ WT,
    const float* __restrict__ bias, const float* __restrict__ Cin,
    float* __restrict__ Out)
{
  __shared__ __attribute__((aligned(16))) __bf16 sA[16 * Ff];   // 4 KB

  const int tid  = threadIdx.x;
  const int wave = tid >> 5;
  const int lane = tid & 31;
  const int row0 = blockIdx.x * 16;
  const int mrow = lane & 15;
  const int hx   = lane >> 4;

  // Cooperative stage: thread -> 8 consecutive k of one row, f32 -> bf16
  {
    const int m = tid >> 4;            // 0..15
    const int k = (tid & 15) * 8;      // 0..120
    const float* ap = A + (size_t)(row0 + m) * Ff + k;
    const float4 f0 = *(const float4*)(ap);
    const float4 f1 = *(const float4*)(ap + 4);
    v8bf pk;
    pk[0]=(__bf16)f0.x; pk[1]=(__bf16)f0.y; pk[2]=(__bf16)f0.z; pk[3]=(__bf16)f0.w;
    pk[4]=(__bf16)f1.x; pk[5]=(__bf16)f1.y; pk[6]=(__bf16)f1.z; pk[7]=(__bf16)f1.w;
    *(v8bf*)(sA + m * Ff + k) = pk;
  }
  __syncthreads();

  const int ncol = wave * 16 + mrow;               // this lane's B column
  const __bf16* wp = WT + (size_t)ncol * Ff;
  const __bf16* arow = sA + mrow * Ff;

  v8f c = {};
  #pragma unroll
  for (int k0 = 0; k0 < Ff; k0 += 32) {
    v8bf a0 = *(const v8bf*)(arow + k0 + 8 * hx);
    v8bf a1 = *(const v8bf*)(arow + k0 + 16 + 8 * hx);
    v16bf af = __builtin_shufflevector(a0, a1,
                 0,1,2,3,4,5,6,7,8,9,10,11,12,13,14,15);
    v8bf b0 = *(const v8bf*)(wp + k0 + 8 * hx);
    v8bf b1 = *(const v8bf*)(wp + k0 + 16 + 8 * hx);
    v16bf bv = __builtin_shufflevector(b0, b1,
                 0,1,2,3,4,5,6,7,8,9,10,11,12,13,14,15);
    c = __builtin_amdgcn_wmma_f32_16x16x32_bf16(false, af, false, bv,
                                                (short)0, c, false, false);
  }

  const float bvs = BIAS ? bias[ncol] : 0.0f;
  #pragma unroll
  for (int j = 0; j < 8; j++) {
    const int m = row0 + 8 * hx + j;
    float v = c[j] + bvs;
    if (CIN)  v += Cin[(size_t)m * Ff + ncol];
    if (RELU) v = fmaxf(v, 0.0f);
    Out[(size_t)m * Ff + ncol] = v;
  }
}

// ---------------------------------------------------------------------------
// Elementwise helpers
// ---------------------------------------------------------------------------
__global__ void k_add(const float* __restrict__ a, const float* __restrict__ b,
                      float* __restrict__ o, int n)
{
  int i = blockIdx.x * 256 + threadIdx.x;
  if (i < n) o[i] = a[i] + b[i];
}

__global__ void k_eig(const float* __restrict__ x, const float* __restrict__ eigvec,
                      const float* __restrict__ eigval, float* __restrict__ o, int n)
{
  int i = blockIdx.x * 256 + threadIdx.x;
  if (i < n) {
    int f  = i & (Ff - 1);
    int nf = i % (Nn * Ff);
    o[i] = x[i] + eigvec[nf] * eigval[f];
  }
}

// Prev[b,t,n,:] = (t>0) ? x[b,t-1,n,:] : 0
__global__ void k_shift(const float* __restrict__ x, float* __restrict__ o, int n)
{
  int i = blockIdx.x * 256 + threadIdx.x;
  if (i < n) {
    int row = i / Ff;
    int t = (row / Nn) % Tt;
    o[i] = (t == 0) ? 0.0f : x[i - Nn * Ff];
  }
}

// out = LayerNorm(y + res) over F=128 (one block per token row)
__global__ __launch_bounds__(128) void k_add_ln(const float* __restrict__ y,
                                                const float* __restrict__ res,
                                                float* __restrict__ o)
{
  const int row = blockIdx.x, f = threadIdx.x;
  __shared__ float red[128];
  float v = y[(size_t)row * Ff + f] + res[(size_t)row * Ff + f];
  red[f] = v; __syncthreads();
  for (int s = 64; s > 0; s >>= 1) { if (f < s) red[f] += red[f + s]; __syncthreads(); }
  float mean = red[0] * (1.0f / Ff); __syncthreads();
  float d = v - mean;
  red[f] = d * d; __syncthreads();
  for (int s = 64; s > 0; s >>= 1) { if (f < s) red[f] += red[f + s]; __syncthreads(); }
  float var = red[0] * (1.0f / Ff);
  o[(size_t)row * Ff + f] = d * rsqrtf(var + 1e-5f);
}

// ---------------------------------------------------------------------------
// Temporal causal attention per (b,n): 12x12 scores, softmax, weighted V
// ---------------------------------------------------------------------------
__global__ __launch_bounds__(256) void k_tatt(const float* __restrict__ Q,
                                              const float* __restrict__ K,
                                              const float* __restrict__ V,
                                              float* __restrict__ Out)
{
  const int bn = blockIdx.x, b = bn / Nn, n = bn % Nn;
  __shared__ float sq[Tt][Ff], sk[Tt][Ff], sv[Tt][Ff];
  __shared__ float att[Tt][Tt];
  const int tid = threadIdx.x;

  for (int i = tid; i < Tt * Ff; i += 256) {
    int t = i / Ff, f = i % Ff;
    size_t idx = (((size_t)b * Tt + t) * Nn + n) * Ff + f;
    sq[t][f] = Q[idx]; sk[t][f] = K[idx]; sv[t][f] = V[idx];
  }
  __syncthreads();

  if (tid < Tt * Tt) {
    int t = tid / Tt, s2 = tid % Tt;
    if (s2 <= t) {
      float acc = 0.0f;
      for (int f = 0; f < Ff; f++) acc += sq[t][f] * sk[s2][f];
      att[t][s2] = acc * INV_SQD;
    } else att[t][s2] = 0.0f;
  }
  __syncthreads();

  if (tid < Tt) {               // masked softmax over s <= t
    int t = tid;
    float mx = -1e30f;
    for (int s2 = 0; s2 <= t; s2++) mx = fmaxf(mx, att[t][s2]);
    float sum = 0.0f;
    for (int s2 = 0; s2 <= t; s2++) { float e = __expf(att[t][s2] - mx); att[t][s2] = e; sum += e; }
    float inv = 1.0f / sum;
    for (int s2 = 0; s2 <= t; s2++) att[t][s2] *= inv;
  }
  __syncthreads();

  for (int i = tid; i < Tt * Ff; i += 256) {
    int t = i / Ff, f = i % Ff;
    float acc = 0.0f;
    for (int s2 = 0; s2 <= t; s2++) acc += att[t][s2] * sv[s2][f];
    Out[(((size_t)b * Tt + t) * Nn + n) * Ff + f] = acc;
  }
}

// ---------------------------------------------------------------------------
// SSA gating score M[b,t,n]: one wave32 per token, 4 features/lane
// ---------------------------------------------------------------------------
__global__ __launch_bounds__(32) void k_ssa_score(const float* __restrict__ Q,
                                                  const float* __restrict__ K,
                                                  const float* __restrict__ V,
                                                  const int* __restrict__ adj,
                                                  const float* __restrict__ pw,
                                                  const float* __restrict__ pb,
                                                  float* __restrict__ Msc)
{
  const int row  = blockIdx.x;
  const int n    = row % Nn;
  const int base = row - n;            // (b,t,0)
  const int lane = threadIdx.x;

  const float* qr = Q + (size_t)row * Ff;
  float q0 = qr[lane], q1 = qr[lane+32], q2 = qr[lane+64], q3 = qr[lane+96];
  float g0 = 0, g1 = 0, g2 = 0, g3 = 0;

  for (int k = 0; k < Kadj; k++) {
    int nb = adj[n * Kadj + k];
    const float* kr = K + (size_t)(base + nb) * Ff;
    float d = q0*kr[lane] + q1*kr[lane+32] + q2*kr[lane+64] + q3*kr[lane+96];
    for (int off = 16; off > 0; off >>= 1) d += __shfl_xor(d, off, 32);
    const float* vr = V + (size_t)(base + nb) * Ff;
    g0 += d*vr[lane]; g1 += d*vr[lane+32]; g2 += d*vr[lane+64]; g3 += d*vr[lane+96];
  }
  float m = g0*pw[lane] + g1*pw[lane+32] + g2*pw[lane+64] + g3*pw[lane+96];
  for (int off = 16; off > 0; off >>= 1) m += __shfl_xor(m, off, 32);
  if (lane == 0) Msc[row] = m + pb[0];
}

// top-36 of 512 per (b,t): iterative block argmax (first-index tie break)
__global__ __launch_bounds__(256) void k_topk(const float* __restrict__ Msc,
                                              int* __restrict__ mtop)
{
  const int bt = blockIdx.x, tid = threadIdx.x;
  __shared__ float sv[Nn];
  __shared__ float rv[256];
  __shared__ int   ri[256];
  sv[tid]       = Msc[(size_t)bt * Nn + tid];
  sv[tid + 256] = Msc[(size_t)bt * Nn + tid + 256];
  __syncthreads();
  for (int s = 0; s < NSAMP; s++) {
    float v0 = sv[tid], v1 = sv[tid + 256];
    ri[tid] = (v0 >= v1) ? tid : tid + 256;
    rv[tid] = fmaxf(v0, v1);
    __syncthreads();
    for (int st = 128; st > 0; st >>= 1) {
      if (tid < st && rv[tid + st] > rv[tid]) { rv[tid] = rv[tid + st]; ri[tid] = ri[tid + st]; }
      __syncthreads();
    }
    if (tid == 0) { mtop[bt * NSAMP + s] = ri[0]; sv[ri[0]] = -3.0e38f; }
    __syncthreads();
  }
}

// Sampled attention per (b,t,s): scores(512), softmax, av(128); store attn row
__global__ __launch_bounds__(256) void k_sattn(const float* __restrict__ Q,
                                               const float* __restrict__ K,
                                               const float* __restrict__ V,
                                               const int* __restrict__ mtop,
                                               float* __restrict__ attn,
                                               float* __restrict__ av)
{
  const int bts = blockIdx.x, bt = bts / NSAMP, s = bts % NSAMP;
  const int tid = threadIdx.x;
  __shared__ float sq[Ff];
  __shared__ float sc[Nn];
  __shared__ float red[256];

  const int qn = mtop[bt * NSAMP + s];
  const int rowbase = bt * Nn;
  if (tid < Ff) sq[tid] = Q[(size_t)(rowbase + qn) * Ff + tid];
  __syncthreads();

  for (int n = tid; n < Nn; n += 256) {
    const float* kr = K + (size_t)(rowbase + n) * Ff;
    float d = 0.0f;
    for (int f = 0; f < Ff; f++) d += sq[f] * kr[f];
    sc[n] = d * INV_SQD;
  }
  __syncthreads();

  red[tid] = fmaxf(sc[tid], sc[tid + 256]); __syncthreads();
  for (int st = 128; st > 0; st >>= 1) { if (tid < st) red[tid] = fmaxf(red[tid], red[tid + st]); __syncthreads(); }
  float mx = red[0]; __syncthreads();

  float e0 = __expf(sc[tid] - mx), e1 = __expf(sc[tid + 256] - mx);
  sc[tid] = e0; sc[tid + 256] = e1;
  red[tid] = e0 + e1; __syncthreads();
  for (int st = 128; st > 0; st >>= 1) { if (tid < st) red[tid] += red[tid + st]; __syncthreads(); }
  float inv = 1.0f / red[0];
  sc[tid] *= inv; sc[tid + 256] *= inv;
  __syncthreads();

  attn[(size_t)bts * Nn + tid]       = sc[tid];
  attn[(size_t)bts * Nn + tid + 256] = sc[tid + 256];
  if (tid < Ff) {
    float acc = 0.0f;
    for (int n = 0; n < Nn; n++) acc += sc[n] * V[(size_t)(rowbase + n) * Ff + tid];
    av[(size_t)bts * Ff + tid] = acc;
  }
}

// cp[n] = argmax_s attn[b,t,s,n]; out[row,:] = av[b,t,cp,:]
__global__ __launch_bounds__(128) void k_cpgather(const float* __restrict__ attn,
                                                  const float* __restrict__ av,
                                                  float* __restrict__ out)
{
  const int row = blockIdx.x, bt = row / Nn, n = row % Nn;
  __shared__ int cp;
  if (threadIdx.x == 0) {
    float best = -1e30f; int bi = 0;
    for (int s = 0; s < NSAMP; s++) {
      float v = attn[((size_t)bt * NSAMP + s) * Nn + n];
      if (v > best) { best = v; bi = s; }
    }
    cp = bi;
  }
  __syncthreads();
  out[(size_t)row * Ff + threadIdx.x] = av[((size_t)bt * NSAMP + cp) * Ff + threadIdx.x];
}

// ---------------------------------------------------------------------------
extern "C" void kernel_launch(void* const* d_in, const int* in_sizes, int n_in,
                              void* d_out, int out_size, void* d_ws, size_t ws_size,
                              hipStream_t stream)
{
  const float* xl     = (const float*)d_in[0];
  const float* xh     = (const float*)d_in[1];
  const float* te     = (const float*)d_in[2];
  const float* eigval = (const float*)d_in[3];
  const float* eigvec = (const float*)d_in[4];
  const int*   adj    = (const int*)d_in[5];
  const float* P[42];
  for (int i = 0; i < 42; i++) P[i] = (const float*)d_in[6 + i];
  // P layout (setup_inputs insertion order):
  //  [0..11]  ta: q_w,q_b,k_w,k_b,v_w,v_b,o_w,o_b,ff_w1,ff_b1,ff_w2,ff_b2
  //  [12..23] sl  [24..35] sh  [36,37] sl_proj_w/b  [38,39] sh_proj_w/b
  //  [40] conv_w (F,F,1,2)  [41] conv_b

  const int total = Mtok * Ff;
  size_t off = 0;
  auto alloc = [&](size_t nbytes) {
    void* p = (char*)d_ws + off;
    off += ((nbytes + 255) & ~(size_t)255);
    return p;
  };
  float* X    = (float*)alloc((size_t)total * 4);   // x+te / x_ / ff out
  float* Qb   = (float*)alloc((size_t)total * 4);
  float* Kb   = (float*)alloc((size_t)total * 4);
  float* Vb   = (float*)alloc((size_t)total * 4);
  float* T1   = (float*)alloc((size_t)total * 4);
  float* T2   = (float*)alloc((size_t)total * 4);
  float* Cur  = (float*)alloc((size_t)total * 4);   // carried residual
  float* Msc  = (float*)alloc((size_t)Mtok * 4);
  int*   mtop = (int*)  alloc((size_t)Bsz * Tt * NSAMP * 4);
  float* attnS= (float*)alloc((size_t)Bsz * Tt * NSAMP * Nn * 4);
  float* avS  = (float*)alloc((size_t)Bsz * Tt * NSAMP * Ff * 4);
  __bf16* WTb = (__bf16*)alloc((size_t)20 * Ff * Ff * 2);   // 20 bf16 weights
  float* outF = (float*)d_out;

  __bf16* WT[20];
  for (int j = 0; j < 20; j++) WT[j] = WTb + (size_t)j * Ff * Ff;

  const int g256  = (total + 255) / 256;
  const int gGemm = Mtok / 16;          // 3072 tile-rows
  const int gW    = (Ff * Ff) / 256;    // 64 blocks per weight

  // ---- Pre-convert all weights to transposed bf16 (WT[n][k]) ----
  {
    int di = 0;
    const int bases[3] = {0, 12, 24};   // ta, sl, sh
    for (int pi = 0; pi < 3; pi++)
      for (int j = 0; j < 6; j++)
        k_wcvt<<<gW, 256, 0, stream>>>(P[bases[pi] + 2 * j], WT[di++], Ff, 1, 0);
    k_wcvt<<<gW, 256, 0, stream>>>(P[40], WT[18], 2, 256, 0);  // conv tap 0 (prev)
    k_wcvt<<<gW, 256, 0, stream>>>(P[40], WT[19], 2, 256, 1);  // conv tap 1 (cur)
  }

  // Epilogue-specialized GEMM launchers
  auto gemm_b = [&](const float* Ap, const __bf16* Wp, const float* bp, float* Op) {
    k_gemm_wmma<1,0,0><<<gGemm, 256, 0, stream>>>(Ap, Wp, bp, nullptr, Op);
  };
  auto gemm_br = [&](const float* Ap, const __bf16* Wp, const float* bp, float* Op) {
    k_gemm_wmma<1,0,1><<<gGemm, 256, 0, stream>>>(Ap, Wp, bp, nullptr, Op);
  };
  auto gemm_cr = [&](const float* Ap, const __bf16* Wp, const float* Cp, float* Op) {
    k_gemm_wmma<0,1,1><<<gGemm, 256, 0, stream>>>(Ap, Wp, nullptr, Cp, Op);
  };
  auto addln = [&](const float* y, const float* r, float* o) {
    k_add_ln<<<Mtok, 128, 0, stream>>>(y, r, o);
  };
  auto run_ssa = [&](const float* Xin, int wtb, int pbase, const float* projw,
                     const float* projb, float* dst) {
    k_eig<<<g256, 256, 0, stream>>>(Xin, eigvec, eigval, X, total);     // X = x_
    gemm_b(X, WT[wtb+0], P[pbase+1], Qb);
    gemm_b(X, WT[wtb+1], P[pbase+3], Kb);
    gemm_b(X, WT[wtb+2], P[pbase+5], Vb);
    k_ssa_score<<<Mtok, 32, 0, stream>>>(Qb, Kb, Vb, adj, projw, projb, Msc);
    k_topk<<<Bsz * Tt, 256, 0, stream>>>(Msc, mtop);
    k_sattn<<<Bsz * Tt * NSAMP, 256, 0, stream>>>(Qb, Kb, Vb, mtop, attnS, avS);
    k_cpgather<<<Mtok, 128, 0, stream>>>(attnS, avS, T1);               // T1 = val
    gemm_b(T1, WT[wtb+3], P[pbase+7], T2);
    addln(T2, X, T1);                                                    // T1 = LN(o+x_)
    gemm_br(T1, WT[wtb+4], P[pbase+9], T2);                              // relu ff1
    gemm_b(T2, WT[wtb+5], P[pbase+11], X);                               // ff2
    addln(X, T1, T2);                                                    // T2 = ssa out
    k_add<<<g256, 256, 0, stream>>>(T2, Xin, dst, total);                // + residual
  };

  // ---- Phase 1: temporal attention on xl ----
  k_add<<<g256, 256, 0, stream>>>(xl, te, X, total);                     // X = xl+te
  gemm_b(X, WT[0], P[1], Qb);
  gemm_b(X, WT[1], P[3], Kb);
  gemm_b(X, WT[2], P[5], Vb);
  k_tatt<<<Bsz * Nn, 256, 0, stream>>>(Qb, Kb, Vb, T1);
  gemm_b(T1, WT[3], P[7], T2);
  addln(T2, X, Cur);                                                     // Cur = LN(o+x)
  gemm_br(Cur, WT[4], P[9], T1);
  gemm_b(T1, WT[5], P[11], T2);
  addln(T2, Cur, Cur);                                                   // Cur = XL1

  // ---- Phase 2: SSA 'sl', writes xl output ----
  run_ssa(Cur, 6, 12, P[36], P[37], outF);

  // ---- Phase 3: causal TCN on xh (two WMMA GEMMs over conv taps) ----
  k_shift<<<g256, 256, 0, stream>>>(xh, T1, total);                      // T1 = x[t-1]
  gemm_b(xh, WT[19], P[41], T2);                                         // W1*x[t]+b
  gemm_cr(T1, WT[18], T2, Cur);                                          // +W0*x[t-1], relu

  // ---- Phase 4: SSA 'sh', writes xh output ----
  run_ssa(Cur, 12, 24, P[38], P[39], outF + total);

  (void)in_sizes; (void)n_in; (void)out_size; (void)ws_size;
}
